// Qwen3VLMoeTextDecoderLayer_76089640616624
// MI455X (gfx1250) — compile-verified, matrix operand order checked by reading, not run on hardware
//
#include <hip/hip_runtime.h>
#include <hip/hip_bf16.h>

// ---------------------------------------------------------------------------
// Qwen3-VL MoE decoder layer for gfx1250 (MI455X), wave32 + WMMA bf16.
// T=2048, D=2048, N_Q=16, N_KV=4, H=128, E=16, F=1024, TOP_K=4, ROT_DIM=128
// ---------------------------------------------------------------------------

#define TT 2048
#define DD 2048
#define NQ 16
#define NKV 4
#define HH 128
#define EE 16
#define FF 1024

typedef __attribute__((ext_vector_type(16))) __bf16 v16bf;
typedef __attribute__((ext_vector_type(8)))  float  v8f;
typedef __attribute__((ext_vector_type(4)))  int    v4i_;
typedef __attribute__((address_space(1))) v4i_* gptr_b128;
typedef __attribute__((address_space(3))) v4i_* lptr_b128;

#if defined(__HIP_DEVICE_COMPILE__) && \
    __has_builtin(__builtin_amdgcn_global_load_async_to_lds_b128) && \
    __has_builtin(__builtin_amdgcn_s_wait_asynccnt)
#define USE_ASYNC_LDS 1
#else
#define USE_ASYNC_LDS 0
#endif

// hardware RNE f32 -> bf16 (v_cvt), not emulation
__device__ __forceinline__ unsigned short f32_to_bf16(float f) {
  union { __bf16 h; unsigned short u; } c;
  c.h = (__bf16)f;
  return c.u;
}
__device__ __forceinline__ unsigned int pack2_bf16(float lo, float hi) {
  union { __bf16 h[2]; unsigned int u; } c;
  c.h[0] = (__bf16)lo; c.h[1] = (__bf16)hi;
  return c.u;
}
__device__ __forceinline__ float bf16_to_f32(unsigned short h) {
  union { unsigned int u; float f; } c; c.u = ((unsigned int)h) << 16; return c.f;
}

// 16-byte global -> LDS copy; async on gfx1250 when the builtin exists.
__device__ __forceinline__ void copy16_g2l(const void* g, void* l) {
#if USE_ASYNC_LDS
  __builtin_amdgcn_global_load_async_to_lds_b128((gptr_b128)g, (lptr_b128)l, 0, 0);
#else
  *(uint4*)l = *(const uint4*)g;
#endif
}
__device__ __forceinline__ void wait_async_all() {
#if USE_ASYNC_LDS
  __builtin_amdgcn_s_wait_asynccnt(0);
#endif
}

// ---------------------------------------------------------------------------
// RMSNorm over last dim, output bf16.   grid = rows, block = 256
// ---------------------------------------------------------------------------
__global__ __launch_bounds__(256) void rmsnorm_to_bf16(
    const float* __restrict__ x, const float* __restrict__ w,
    unsigned short* __restrict__ out, int D) {
  int row = blockIdx.x;
  const float* xr = x + (size_t)row * D;
  float ss = 0.f;
  for (int d = threadIdx.x; d < D; d += 256) { float v = xr[d]; ss += v * v; }
  for (int o = 16; o > 0; o >>= 1) ss += __shfl_xor(ss, o, 32);
  __shared__ float red[8];
  if ((threadIdx.x & 31) == 0) red[threadIdx.x >> 5] = ss;
  __syncthreads();
  if (threadIdx.x < 8) {
    float v = red[threadIdx.x];
    for (int o = 4; o > 0; o >>= 1) v += __shfl_xor(v, o, 32);
    if (threadIdx.x == 0) red[0] = v;
  }
  __syncthreads();
  float inv = rsqrtf(red[0] / (float)D + 1e-6f);
  for (int d = threadIdx.x; d < D; d += 256)
    out[(size_t)row * D + d] = f32_to_bf16(xr[d] * inv * w[d]);
}

// ---------------------------------------------------------------------------
// Per-head RMSNorm + RoPE (rot_dim == H == 128), output bf16.
// grid = (T, nh), block = 128
// ---------------------------------------------------------------------------
__global__ __launch_bounds__(128) void qknorm_rope(
    const float* __restrict__ in, const float* __restrict__ nw,
    const float* __restrict__ cosb, const float* __restrict__ sinb,
    unsigned short* __restrict__ out, int nh) {
  int t = blockIdx.x, n = blockIdx.y, h = threadIdx.x;
  const float* r = in + ((size_t)t * nh + n) * HH;
  float v = r[h];
  float ss = v * v;
  for (int o = 16; o > 0; o >>= 1) ss += __shfl_xor(ss, o, 32);
  __shared__ float red[4];
  if ((h & 31) == 0) red[h >> 5] = ss;
  __syncthreads();
  float tot = red[0] + red[1] + red[2] + red[3];
  float x = v * rsqrtf(tot / (float)HH + 1e-6f) * nw[h];
  __shared__ float sh[HH];
  sh[h] = x;
  __syncthreads();
  float c = cosb[(size_t)t * HH + h], s = sinb[(size_t)t * HH + h];
  float o;
  if (h < 64) o = x * c - sh[h + 64] * s;
  else        o = x * c + sh[h - 64] * s;
  out[((size_t)t * nh + n) * HH + h] = f32_to_bf16(o);
}

// ---------------------------------------------------------------------------
// Row softmax (f32 in, bf16 out).  grid = rows, block = 256
// ---------------------------------------------------------------------------
__global__ __launch_bounds__(256) void softmax_row_bf16(
    const float* __restrict__ S, unsigned short* __restrict__ P, int N) {
  int row = blockIdx.x;
  const float* r = S + (size_t)row * N;
  __shared__ float red[8];
  float mx = -1e30f;
  for (int i = threadIdx.x; i < N; i += 256) mx = fmaxf(mx, r[i]);
  for (int o = 16; o > 0; o >>= 1) mx = fmaxf(mx, __shfl_xor(mx, o, 32));
  if ((threadIdx.x & 31) == 0) red[threadIdx.x >> 5] = mx;
  __syncthreads();
  if (threadIdx.x < 8) {
    float v = red[threadIdx.x];
    for (int o = 4; o > 0; o >>= 1) v = fmaxf(v, __shfl_xor(v, o, 32));
    if (threadIdx.x == 0) red[0] = v;
  }
  __syncthreads();
  mx = red[0];
  __syncthreads();
  float sum = 0.f;
  for (int i = threadIdx.x; i < N; i += 256) sum += __expf(r[i] - mx);
  for (int o = 16; o > 0; o >>= 1) sum += __shfl_xor(sum, o, 32);
  if ((threadIdx.x & 31) == 0) red[threadIdx.x >> 5] = sum;
  __syncthreads();
  if (threadIdx.x < 8) {
    float v = red[threadIdx.x];
    for (int o = 4; o > 0; o >>= 1) v += __shfl_xor(v, o, 32);
    if (threadIdx.x == 0) red[0] = v;
  }
  __syncthreads();
  float inv = 1.f / red[0];
  for (int i = threadIdx.x; i < N; i += 256)
    P[(size_t)row * N + i] = f32_to_bf16(__expf(r[i] - mx) * inv);
}

// ---------------------------------------------------------------------------
// out = a + b (elementwise f32)
// ---------------------------------------------------------------------------
__global__ __launch_bounds__(256) void add2(
    const float* __restrict__ a, const float* __restrict__ b,
    float* __restrict__ o, int n) {
  int i = blockIdx.x * 256 + threadIdx.x;
  if (i < n) o[i] = a[i] + b[i];
}

// ---------------------------------------------------------------------------
// Router logits: x(bf16 T x D) @ gate_w(D x 16).   grid = T, block = 256
// ---------------------------------------------------------------------------
__global__ __launch_bounds__(256) void router_gemm(
    const unsigned short* __restrict__ x, const float* __restrict__ gw,
    float* __restrict__ logits, int D) {
  int t = blockIdx.x;
  int e = threadIdx.x & 15, part = threadIdx.x >> 4;
  float acc = 0.f;
  for (int d = part; d < D; d += 16)
    acc += bf16_to_f32(x[(size_t)t * D + d]) * gw[(size_t)d * EE + e];
  __shared__ float buf[256];
  buf[threadIdx.x] = acc;
  __syncthreads();
  if (threadIdx.x < 16) {
    float s = 0.f;
    for (int j = 0; j < 16; j++) s += buf[threadIdx.x + j * 16];
    logits[(size_t)t * EE + threadIdx.x] = s;
  }
}

__global__ void zero_i32(int* p, int n) {
  int i = blockIdx.x * blockDim.x + threadIdx.x;
  if (i < n) p[i] = 0;
}

// ---------------------------------------------------------------------------
// Softmax over 16 experts + top-4 + renormalize + per-expert compaction lists.
// ---------------------------------------------------------------------------
__global__ __launch_bounds__(256) void route_topk(
    const float* __restrict__ logits, int* __restrict__ counts,
    int* __restrict__ toks, float* __restrict__ wts, int T) {
  int t = blockIdx.x * 256 + threadIdx.x;
  if (t >= T) return;
  float l[EE]; float mx = -1e30f;
  for (int e = 0; e < EE; e++) { l[e] = logits[(size_t)t * EE + e]; mx = fmaxf(mx, l[e]); }
  float sum = 0.f;
  for (int e = 0; e < EE; e++) { l[e] = __expf(l[e] - mx); sum += l[e]; }
  float invs = 1.f / sum;
  for (int e = 0; e < EE; e++) l[e] *= invs;
  int idx[4]; float w[4]; float tot = 0.f;
  for (int s = 0; s < 4; s++) {
    int bi = 0; float bv = -1.f;
    for (int e = 0; e < EE; e++) if (l[e] > bv) { bv = l[e]; bi = e; }
    idx[s] = bi; w[s] = bv; l[bi] = -2.f; tot += bv;
  }
  float invt = 1.f / tot;
  for (int s = 0; s < 4; s++) {
    int e = idx[s];
    int pos = atomicAdd(&counts[e], 1);
    toks[(size_t)e * T + pos] = t;
    wts[(size_t)e * T + pos] = w[s] * invt;
  }
}

// ---------------------------------------------------------------------------
// gated = up * silu(gate) from gate_up buffer (rows compacted, count on dev).
// ---------------------------------------------------------------------------
__global__ __launch_bounds__(256) void silu_mul(
    const float* __restrict__ gu, const int* __restrict__ cnt,
    unsigned short* __restrict__ gated) {
  int i = blockIdx.x * 256 + threadIdx.x;
  int total = (*cnt) * FF;
  if (i >= total) return;
  int row = i >> 10, f = i & (FF - 1);
  float g = gu[(size_t)row * (2 * FF) + f];
  float u = gu[(size_t)row * (2 * FF) + FF + f];
  float sg = g / (1.f + __expf(-g));
  gated[(size_t)row * FF + f] = f32_to_bf16(u * sg);
}

// ---------------------------------------------------------------------------
// WMMA bf16 GEMM, 128x128x32 tiles, double-buffered LDS, async A/B staging.
//   EPI 0: C f32 store       EPI 1: C bf16 store
//   EPI 2: C = C*scale + mask[m*ldmask+n] (f32)
//   EPI 3: scatter: sout[gidx[m]*ldsout+n] += gw[m]*C   (MoE down)
//   GATHER_A: A row m comes from gidx[m]; row count limited by *gcnt
//   BTRANS: B element (k,n) = B[n*ldb + k] (bf16 src); else B[k*ldb + n] (f32)
// Block: 256 threads = 8 waves; wave computes 32x64 of C (8 WMMAs / K-step).
// ---------------------------------------------------------------------------
template <int EPI, bool GATHER_A, typename TB, bool BTRANS>
__global__ __launch_bounds__(256) void gemm_wmma_bf16(
    const unsigned short* __restrict__ A, int lda,
    const TB* __restrict__ B, int ldb,
    float* __restrict__ C, int ldc,
    int M, int N, int K,
    float scale, const float* __restrict__ mask, int ldmask,
    const int* __restrict__ gidx, const int* __restrict__ gcnt,
    const float* __restrict__ gw, float* __restrict__ sout, int ldsout) {
  constexpr int BM = 128, BN = 128, BK = 32;
  int by = blockIdx.y, bx = blockIdx.x;
  int mlimit = M;
  if (gcnt) mlimit = *gcnt;
  if (by * BM >= mlimit) return;

  __shared__ __align__(16) unsigned short As[2][BM][BK];
  __shared__ __align__(16) unsigned short Bs[2][BN][BK];

  int tid = threadIdx.x;
  int lane = tid & 31, wid = tid >> 5;
  int wm = wid & 3, wn = wid >> 2;     // wave sub-tile: rows wm*32, cols wn*64

  // --- A staging: thread covers (row = tid>>1, 16 consecutive k) ---
  int a_row = tid >> 1, a_kc = (tid & 1) * 16;
  int a_grow = by * BM + a_row;
  int a_src = 0;     // clamp invalid rows to 0; their C rows are discarded
  if (a_grow < mlimit) a_src = GATHER_A ? gidx[a_grow] : a_grow;
  const unsigned short* Arow = A + (size_t)a_src * lda + a_kc;

  // --- B staging assignments ---
  int bt_n = tid >> 1, bt_kc = (tid & 1) * 16;          // BTRANS (bf16 rows)
  int bn_k2 = (tid & 15) * 2, bn_nc = (tid >> 4) * 8;   // !BTRANS (f32 rows)

  auto stageA = [&](int kt, int buf) {
    const unsigned short* g = Arow + kt * BK;
    copy16_g2l(g,     &As[buf][a_row][a_kc]);
    copy16_g2l(g + 8, &As[buf][a_row][a_kc + 8]);
  };
  auto stageB_trans = [&](int kt, int buf) {
    const TB* g = B + (size_t)(bx * BN + bt_n) * ldb + kt * BK + bt_kc;
    copy16_g2l(g,     &Bs[buf][bt_n][bt_kc]);
    copy16_g2l(g + 8, &Bs[buf][bt_n][bt_kc + 8]);
  };

  float br0[8], br1[8];
  auto loadB_ntrans = [&](int kt) {
    const float* p0 = (const float*)B + (size_t)(kt * BK + bn_k2) * ldb + bx * BN + bn_nc;
    const float* p1 = p0 + ldb;
#pragma unroll
    for (int j = 0; j < 8; j++) { br0[j] = p0[j]; br1[j] = p1[j]; }
  };
  auto storeB_ntrans = [&](int buf) {
#pragma unroll
    for (int j = 0; j < 8; j++)
      *(unsigned int*)&Bs[buf][bn_nc + j][bn_k2] = pack2_bf16(br0[j], br1[j]);
  };

  v8f acc[2][4];
  v8f zz = {};
#pragma unroll
  for (int i = 0; i < 2; i++)
#pragma unroll
    for (int j = 0; j < 4; j++) acc[i][j] = zz;

  int l15 = lane & 15, kb = (lane >> 4) * 8;

  auto compute = [&](int buf) {
    v16bf a0, a1;
    {
      union { uint4 q[2]; v16bf v; } u;
      u.q[0] = *(const uint4*)&As[buf][wm * 32 + l15][kb];
      u.q[1] = *(const uint4*)&As[buf][wm * 32 + l15][kb + 16];
      a0 = u.v;
    }
    {
      union { uint4 q[2]; v16bf v; } u;
      u.q[0] = *(const uint4*)&As[buf][wm * 32 + 16 + l15][kb];
      u.q[1] = *(const uint4*)&As[buf][wm * 32 + 16 + l15][kb + 16];
      a1 = u.v;
    }
#pragma unroll
    for (int j = 0; j < 4; j++) {
      v16bf bj;
      union { uint4 q[2]; v16bf v; } u;
      u.q[0] = *(const uint4*)&Bs[buf][wn * 64 + j * 16 + l15][kb];
      u.q[1] = *(const uint4*)&Bs[buf][wn * 64 + j * 16 + l15][kb + 16];
      bj = u.v;
      acc[0][j] = __builtin_amdgcn_wmma_f32_16x16x32_bf16(false, a0, false, bj,
                                                          (short)0, acc[0][j], false, false);
      acc[1][j] = __builtin_amdgcn_wmma_f32_16x16x32_bf16(false, a1, false, bj,
                                                          (short)0, acc[1][j], false, false);
    }
  };

  // --- software-pipelined K loop with double-buffered LDS ---
  int nk = K / BK;
  stageA(0, 0);
  if constexpr (BTRANS) stageB_trans(0, 0);
  else { loadB_ntrans(0); storeB_ntrans(0); }
  wait_async_all();
  __syncthreads();

  for (int kt = 0; kt < nk; ++kt) {
    int cur = kt & 1, nxt = cur ^ 1;
    bool have_next = (kt + 1 < nk);
    if (have_next) {
      stageA(kt + 1, nxt);
      if constexpr (BTRANS) stageB_trans(kt + 1, nxt);
      else loadB_ntrans(kt + 1);
    }
    compute(cur);
    if constexpr (!BTRANS) { if (have_next) storeB_ntrans(nxt); }
    wait_async_all();
    __syncthreads();
  }

  // --- epilogue (C 16x16 f32 layout: VGPR r -> M = r + 8*(lane>=16), N = lane&15)
  int lm8 = (lane >> 4) * 8;
#pragma unroll
  for (int i = 0; i < 2; i++) {
#pragma unroll
    for (int j = 0; j < 4; j++) {
      int n = bx * BN + wn * 64 + j * 16 + l15;
      if (n >= N) continue;
#pragma unroll
      for (int r = 0; r < 8; r++) {
        int m = by * BM + wm * 32 + i * 16 + lm8 + r;
        if (m >= mlimit) continue;
        float v = acc[i][j][r];
        if constexpr (EPI == 0) {
          C[(size_t)m * ldc + n] = v;
        } else if constexpr (EPI == 1) {
          ((unsigned short*)C)[(size_t)m * ldc + n] = f32_to_bf16(v);
        } else if constexpr (EPI == 2) {
          C[(size_t)m * ldc + n] = v * scale + mask[(size_t)m * ldmask + n];
        } else {  // EPI == 3: weighted scatter-add
          int t = gidx[m];
          float w = gw[m];
          sout[(size_t)t * ldsout + n] += w * v;
        }
      }
    }
  }
}

// ---------------------------------------------------------------------------
// Host orchestration
// ---------------------------------------------------------------------------
extern "C" void kernel_launch(void* const* d_in, const int* in_sizes, int n_in,
                              void* d_out, int out_size, void* d_ws, size_t ws_size,
                              hipStream_t stream) {
  const float* hidden   = (const float*)d_in[0];
  const float* cosb     = (const float*)d_in[1];
  const float* sinb     = (const float*)d_in[2];
  const float* mask     = (const float*)d_in[3];   // (T,1,T) flat T*T
  const float* in_ln    = (const float*)d_in[4];
  const float* post_ln  = (const float*)d_in[5];
  const float* q_w      = (const float*)d_in[6];   // D x 2048
  const float* k_w      = (const float*)d_in[7];   // D x 512
  const float* v_w      = (const float*)d_in[8];   // D x 512
  const float* o_w      = (const float*)d_in[9];   // 2048 x D
  const float* q_norm_w = (const float*)d_in[10];
  const float* k_norm_w = (const float*)d_in[11];
  const float* gate_w   = (const float*)d_in[12];  // D x 16
  const float* gate_up_w= (const float*)d_in[13];  // E x D x 2048
  const float* down_w   = (const float*)d_in[14];  // E x 1024 x D
  float* out = (float*)d_out;                      // T x D

  // --- bump allocator over workspace ---
  char* ws = (char*)d_ws;
  auto alloc = [&](size_t bytes) {
    void* p = (void*)ws;
    ws += (bytes + 255) & ~(size_t)255;
    return p;
  };
  const size_t T = TT, D = DD;
  unsigned short* xn   = (unsigned short*)alloc(T * D * 2);         // ln1(x) bf16
  float*          qf   = (float*)alloc(T * 2048 * 4);               // q proj f32
  float*          kf   = (float*)alloc(T * 512 * 4);
  float*          vf   = (float*)alloc(T * 512 * 4);
  unsigned short* qb   = (unsigned short*)alloc(T * 2048 * 2);      // q post rope bf16
  unsigned short* kbuf = (unsigned short*)alloc(T * 512 * 2);       // k post rope bf16
  float*          Sbuf = (float*)alloc(T * T * 4);                  // scores f32
  unsigned short* Pbuf = (unsigned short*)alloc(T * T * 2);         // probs bf16
  unsigned short* aob  = (unsigned short*)alloc(T * 2048 * 2);      // attn out bf16
  float*          oo   = (float*)alloc(T * D * 4);                  // o proj f32
  unsigned short* xn2  = (unsigned short*)alloc(T * D * 2);         // ln2(h) bf16
  float*          lgts = (float*)alloc(T * EE * 4);
  int*            cnts = (int*)alloc(EE * 4);
  int*            toks = (int*)alloc((size_t)EE * T * 4);
  float*          wts  = (float*)alloc((size_t)EE * T * 4);
  float*          ge   = (float*)alloc(T * 2048 * 4);               // gate_up out f32
  unsigned short* gated= (unsigned short*)alloc(T * FF * 2);        // silu-gated bf16

  auto nil_i = (const int*)nullptr;
  auto nil_f = (const float*)nullptr;
  auto g2 = [](int n, int m) { return dim3((n + 127) / 128, (m + 127) / 128); };

  // 1) input layernorm -> bf16
  rmsnorm_to_bf16<<<(int)T, 256, 0, stream>>>(hidden, in_ln, xn, (int)D);

  // 2) QKV projections (WMMA bf16)
  gemm_wmma_bf16<0, false, float, false><<<g2(2048, TT), 256, 0, stream>>>(
      xn, DD, q_w, 2048, qf, 2048, TT, 2048, DD, 1.f, nil_f, 0, nil_i, nil_i, nil_f, nullptr, 0);
  gemm_wmma_bf16<0, false, float, false><<<g2(512, TT), 256, 0, stream>>>(
      xn, DD, k_w, 512, kf, 512, TT, 512, DD, 1.f, nil_f, 0, nil_i, nil_i, nil_f, nullptr, 0);
  gemm_wmma_bf16<0, false, float, false><<<g2(512, TT), 256, 0, stream>>>(
      xn, DD, v_w, 512, vf, 512, TT, 512, DD, 1.f, nil_f, 0, nil_i, nil_i, nil_f, nullptr, 0);

  // 3) per-head q/k norm + rope -> bf16
  qknorm_rope<<<dim3(TT, NQ), 128, 0, stream>>>(qf, q_norm_w, cosb, sinb, qb, NQ);
  qknorm_rope<<<dim3(TT, NKV), 128, 0, stream>>>(kf, k_norm_w, cosb, sinb, kbuf, NKV);

  // 4) attention per query head (GQA: kv head = qh/4)
  const float scaling = 0.08838834764831845f;  // 128^-0.5
  for (int qh = 0; qh < NQ; ++qh) {
    int kvh = qh / (NQ / NKV);
    // S = Q Kt * scale + mask
    gemm_wmma_bf16<2, false, unsigned short, true><<<g2(TT, TT), 256, 0, stream>>>(
        qb + (size_t)qh * HH, 2048, kbuf + (size_t)kvh * HH, 512,
        Sbuf, TT, TT, TT, HH, scaling, mask, TT, nil_i, nil_i, nil_f, nullptr, 0);
    // P = softmax(S) -> bf16
    softmax_row_bf16<<<TT, 256, 0, stream>>>(Sbuf, Pbuf, TT);
    // O_h = P V_h  -> bf16 into attn-out column block
    gemm_wmma_bf16<1, false, float, false><<<g2(HH, TT), 256, 0, stream>>>(
        Pbuf, TT, vf + (size_t)kvh * HH, 512,
        (float*)(aob + (size_t)qh * HH), 2048, TT, HH, TT, 1.f,
        nil_f, 0, nil_i, nil_i, nil_f, nullptr, 0);
  }

  // 5) output projection + residual
  gemm_wmma_bf16<0, false, float, false><<<g2(DD, TT), 256, 0, stream>>>(
      aob, 2048, o_w, DD, oo, DD, TT, DD, 2048, 1.f, nil_f, 0, nil_i, nil_i, nil_f, nullptr, 0);
  add2<<<(TT * DD) / 256, 256, 0, stream>>>(hidden, oo, out, TT * DD);

  // 6) post layernorm -> bf16
  rmsnorm_to_bf16<<<TT, 256, 0, stream>>>(out, post_ln, xn2, DD);

  // 7) routing
  router_gemm<<<TT, 256, 0, stream>>>(xn2, gate_w, lgts, DD);
  zero_i32<<<1, 32, 0, stream>>>(cnts, EE);
  route_topk<<<TT / 256, 256, 0, stream>>>(lgts, cnts, toks, wts, TT);

  // 8) MoE experts (gathered GEMMs; d_out already holds residual, scatter adds)
  for (int e = 0; e < EE; ++e) {
    const float* guw = gate_up_w + (size_t)e * DD * 2048;
    const float* dnw = down_w + (size_t)e * FF * DD;
    const int* tl = toks + (size_t)e * TT;
    const float* wl = wts + (size_t)e * TT;
    // gate_up: gathered rows of xn2, N = 2F = 2048
    gemm_wmma_bf16<0, true, float, false><<<g2(2048, TT), 256, 0, stream>>>(
        xn2, DD, guw, 2048, ge, 2048, TT, 2048, DD, 1.f,
        nil_f, 0, tl, cnts + e, nil_f, nullptr, 0);
    // gated = up * silu(gate)
    silu_mul<<<(TT * FF) / 256, 256, 0, stream>>>(ge, cnts + e, gated);
    // down + weighted scatter-add into out
    gemm_wmma_bf16<3, false, float, false><<<g2(DD, TT), 256, 0, stream>>>(
        gated, FF, dnw, DD, out, DD, TT, DD, FF, 1.f,
        nil_f, 0, tl, cnts + e, wl, out, DD);
  }
}
